// GIBGIN_25872882991417
// MI455X (gfx1250) — matrix operand order repaired and checked by reading, not compiled, not run on hardware
//
#include <hip/hip_runtime.h>

// ---------------------------------------------------------------------------
// Problem constants (match reference)
// ---------------------------------------------------------------------------
#define NG    512          // graphs
#define NPG   128          // nodes per graph
#define NTOT  (NG * NPG)   // 65536 nodes
#define DEG   16
#define ETOT  (NTOT * DEG) // 1048576 edges
#define EPG   (ETOT / NG)  // 2048 edges per graph
#define H     128
#define KCB   128          // codebook size
#define NCLS  10
#define BN_EPS 1e-5f

using bf_t = __bf16;
typedef __attribute__((ext_vector_type(16))) __bf16 v16bf;
typedef __attribute__((ext_vector_type(8)))  __bf16 v8bf;
typedef __attribute__((ext_vector_type(8)))  float  v8f;

// ---------------------------------------------------------------------------
// Fragment-order LDS layout for v_wmma_f32_16x16x32_bf16 (wave32).
// A tile 16x32 per (m-tile,kc):  K bits [4]=chunk, [3]=lane-half, [2:0]=elem
// B tile 32x16 per (n-tile,kc):  K bits [4]=chunk, [3]=lane-half, [2:0]=elem
// Chunk storage: buf[((t*2+ch)*32 + lane)*8 + i], t = tile*4 + kc/32.
// Each fragment load = two contiguous 16B ds_load_b128 per lane.
// ---------------------------------------------------------------------------
__device__ __forceinline__ v16bf load_frag(const bf_t* buf, int t, int lane) {
  const bf_t* p = buf + ((t * 2) * 32 + lane) * 8;
  v8bf lo = *(const v8bf*)p;
  v8bf hi = *(const v8bf*)(p + 256);   // chunk 1 of this tile
  v16bf r;
#pragma unroll
  for (int i = 0; i < 8; ++i) { r[i] = lo[i]; r[8 + i] = hi[i]; }
  return r;
}

// ---------------------------------------------------------------------------
// Generic 128-wide GEMM: OUT[rows x 128] = act(A_bf16[rows x 128] @ W + bias)
// ACT: 0=none, 1=relu, 2=tanh.  BT: read W transposed (W[c*128+r]).
// BIAS/OUTF/OUTB compile-time flags -> branch-free epilogue.
// One 128x128 output tile per block; 8 waves, each owns one 16-row m-tile.
// ---------------------------------------------------------------------------
template <int ACT, bool BT, bool BIAS, bool OUTF, bool OUTB>
__global__ __launch_bounds__(256) void gemm128_kernel(
    const bf_t* __restrict__ A, const float* __restrict__ W,
    const float* __restrict__ bias, float* __restrict__ outf,
    bf_t* __restrict__ outb) {
  __shared__ bf_t sAf[H * H];   // A in fragment order (32 KB)
  __shared__ bf_t sWf[H * H];   // W in fragment order (32 KB)
  const int tid  = threadIdx.x;
  const int wave = tid >> 5;
  const int lane = tid & 31;
  const int half = lane >> 4;
  const int l    = lane & 15;
  const long row0 = (long)blockIdx.x * 128;

  // A tile: coalesced 16B global loads; each 8-elem K chunk is one fragment
  // chunk -> single 16B swizzled LDS store.
  for (int idx = tid * 8; idx < H * H; idx += 256 * 8) {
    v8bf v = *(const v8bf*)(A + row0 * H + idx);
    const int row  = idx >> 7;
    const int q    = (idx >> 3) & 15;        // K chunk id within row
    const int hb   = q & 1;                  // K bit [3] -> lane half
    const int ch   = (q >> 1) & 1;           // K bit [4] -> chunk
    const int kcid = q >> 2;
    const int t    = (row >> 4) * 4 + kcid;
    *(v8bf*)(sAf + ((t * 2 + ch) * 32 + hb * 16 + (row & 15)) * 8) = v;
  }
  // W: coalesced f32 reads, downconvert, scatter into fragment order (once).
  for (int idx = tid; idx < H * H; idx += 256) {
    const int k = idx >> 7, c = idx & 127;
    const float w = BT ? W[(long)c * H + k] : W[idx];
    const int hb = (k >> 3) & 1;             // K bit [3] -> lane half
    const int ch = (k >> 4) & 1;             // K bit [4] -> chunk
    const int t  = (c >> 4) * 4 + (k >> 5);
    sWf[((t * 2 + ch) * 32 + hb * 16 + (c & 15)) * 8 + (k & 7)] = (bf_t)w;
  }
  __syncthreads();

  // Hoist the 4 A fragments for this wave's m-tile (reused by all 8 n-tiles).
  v16bf afr[4];
#pragma unroll
  for (int kk = 0; kk < 4; ++kk) afr[kk] = load_frag(sAf, wave * 4 + kk, lane);

  const int m0 = wave * 16;
#pragma unroll
  for (int nt = 0; nt < 8; ++nt) {
    v8f acc = {};
#pragma unroll
    for (int kk = 0; kk < 4; ++kk) {
      v16bf bfr = load_frag(sWf, nt * 4 + kk, lane);
      acc = __builtin_amdgcn_wmma_f32_16x16x32_bf16(
          false, afr[kk], false, bfr, (short)0, acc, false, false);
    }
    const int col = nt * 16 + l;
    const float bv = BIAS ? bias[col] : 0.0f;
#pragma unroll
    for (int r = 0; r < 8; ++r) {
      const int row = m0 + half * 8 + r;
      float v = acc[r] + bv;
      if (ACT == 1) v = v > 0.0f ? v : 0.0f;
      else if (ACT == 2) v = tanhf(v);
      const long o = (row0 + row) * H + col;
      if (OUTF) outf[o] = v;
      if (OUTB) outb[o] = (bf_t)v;
    }
  }
}

// ---------------------------------------------------------------------------
// GIN aggregation per graph: t[i] = h[i] + sum_{(s,d) edges, d==i} h[s]
// Accumulates in LDS with ds_add_f32 atomics, emits bf16 for WMMA GEMMs.
// ---------------------------------------------------------------------------
__global__ __launch_bounds__(256) void aggregate_kernel(
    const float* __restrict__ h, const int* __restrict__ ei,
    bf_t* __restrict__ tb) {
  __shared__ float acc[NPG * H];
  const int g = blockIdx.x, tid = threadIdx.x;
  const long base = (long)g * NPG * H;
  for (int idx = tid; idx < NPG * H; idx += 256) acc[idx] = h[base + idx];
  __syncthreads();
  const int wave = tid >> 5, lane = tid & 31;
  const int e0 = g * EPG;
  for (int e = wave; e < EPG; e += 8) {
    const int src = ei[e0 + e];
    const int dl  = ei[ETOT + e0 + e] - g * NPG;  // local dst
    const float* hs = h + (long)src * H;
#pragma unroll
    for (int j = 0; j < 4; ++j) {
      const int f = lane + 32 * j;
      atomicAdd(&acc[dl * H + f], hs[f]);  // ds_add_f32
    }
  }
  __syncthreads();
  for (int idx = tid; idx < NPG * H; idx += 256) tb[base + idx] = (bf_t)acc[idx];
}

// ---------------------------------------------------------------------------
// BatchNorm (training-mode batch stats over all N nodes)
// ---------------------------------------------------------------------------
__global__ __launch_bounds__(256) void bn_stats_kernel(
    const float* __restrict__ v, float* __restrict__ mean,
    float* __restrict__ rstd) {
  const int f = blockIdx.x;
  float s = 0.0f, s2 = 0.0f;
  for (int r = threadIdx.x; r < NTOT; r += 256) {
    const float x = v[(long)r * H + f];
    s += x; s2 += x * x;
  }
  for (int o = 16; o > 0; o >>= 1) {
    s  += __shfl_down(s, o, 32);
    s2 += __shfl_down(s2, o, 32);
  }
  __shared__ float sh[16];
  const int wave = threadIdx.x >> 5, lane = threadIdx.x & 31;
  if (lane == 0) { sh[wave] = s; sh[8 + wave] = s2; }
  __syncthreads();
  if (threadIdx.x == 0) {
    float S = 0.0f, S2 = 0.0f;
    for (int w = 0; w < 8; ++w) { S += sh[w]; S2 += sh[8 + w]; }
    const float mu = S / (float)NTOT;
    mean[f] = mu;
    rstd[f] = rsqrtf(S2 / (float)NTOT - mu * mu + BN_EPS);
  }
}

__global__ __launch_bounds__(256) void bn_apply_kernel(
    const float* __restrict__ v, const float* __restrict__ mean,
    const float* __restrict__ rstd, const float* __restrict__ gamma,
    const float* __restrict__ beta, float* __restrict__ hout,
    bf_t* __restrict__ hb) {
  const long i = (long)blockIdx.x * 256 + threadIdx.x;
  const int f = (int)(i & (H - 1));
  const float x = (v[i] - mean[f]) * rstd[f] * gamma[f] + beta[f];
  hout[i] = x;
  hb[i] = (bf_t)x;
}

// ---------------------------------------------------------------------------
// Assignment a = softmax_2(m @ c2W + c2b); one wave per row
// ---------------------------------------------------------------------------
__global__ __launch_bounds__(256) void assign_kernel(
    const float* __restrict__ m, const float* __restrict__ c2W,
    const float* __restrict__ c2b, float* __restrict__ a) {
  const int wave = threadIdx.x >> 5, lane = threadIdx.x & 31;
  const long row = (long)blockIdx.x * 8 + wave;
  const float* mr = m + row * H;
  float s0 = 0.0f, s1 = 0.0f;
#pragma unroll
  for (int j = 0; j < 4; ++j) {
    const int f = lane + 32 * j;
    const float mv = mr[f];
    s0 += mv * c2W[f * 2 + 0];
    s1 += mv * c2W[f * 2 + 1];
  }
  for (int o = 16; o > 0; o >>= 1) {
    s0 += __shfl_down(s0, o, 32);
    s1 += __shfl_down(s1, o, 32);
  }
  if (lane == 0) {
    s0 += c2b[0]; s1 += c2b[1];
    const float mx = fmaxf(s0, s1);
    const float e0 = __expf(s0 - mx), e1 = __expf(s1 - mx);
    const float inv = 1.0f / (e0 + e1);
    a[row * 2 + 0] = e0 * inv;
    a[row * 2 + 1] = e1 * inv;
  }
}

// pos[g] = sum_i a[i,0]*h[i,:]  ;  colsum[f] += sum_i h[i,f]  (for graph_emb)
__global__ __launch_bounds__(128) void graph_kernel(
    const float* __restrict__ h, const float* __restrict__ a,
    float* __restrict__ pos, float* __restrict__ colsum) {
  const int g = blockIdx.x, f = threadIdx.x;
  float p = 0.0f, cs = 0.0f;
  for (int n = 0; n < NPG; ++n) {
    const long node = (long)g * NPG + n;
    const float hv = h[node * H + f];
    p  += a[node * 2] * hv;
    cs += hv;
  }
  pos[(long)g * H + f] = p;
  atomicAdd(&colsum[f], cs);
}

// new_adj = a^T A a per graph -> L1 row-normalize -> diag penalty
__global__ __launch_bounds__(256) void adj_kernel(
    const float* __restrict__ a, const int* __restrict__ ei,
    float* __restrict__ penalty) {
  const int g = blockIdx.x, tid = threadIdx.x;
  float m00 = 0.0f, m01 = 0.0f, m10 = 0.0f, m11 = 0.0f;
  const int e0 = g * EPG;
  for (int e = tid; e < EPG; e += 256) {
    const int s = ei[e0 + e], d = ei[ETOT + e0 + e];
    const float a0 = a[s * 2], a1 = a[s * 2 + 1];
    const float b0 = a[d * 2], b1 = a[d * 2 + 1];
    m00 += a0 * b0; m01 += a0 * b1; m10 += a1 * b0; m11 += a1 * b1;
  }
  for (int o = 16; o > 0; o >>= 1) {
    m00 += __shfl_down(m00, o, 32); m01 += __shfl_down(m01, o, 32);
    m10 += __shfl_down(m10, o, 32); m11 += __shfl_down(m11, o, 32);
  }
  __shared__ float sm[8][4];
  const int wave = tid >> 5, lane = tid & 31;
  if (lane == 0) { sm[wave][0] = m00; sm[wave][1] = m01; sm[wave][2] = m10; sm[wave][3] = m11; }
  __syncthreads();
  if (tid == 0) {
    float t00 = 0.0f, t01 = 0.0f, t10 = 0.0f, t11 = 0.0f;
    for (int w = 0; w < 8; ++w) { t00 += sm[w][0]; t01 += sm[w][1]; t10 += sm[w][2]; t11 += sm[w][3]; }
    const float d0 = t00 / fmaxf(fabsf(t00) + fabsf(t01), 1e-12f);
    const float d1 = t11 / fmaxf(fabsf(t10) + fabsf(t11), 1e-12f);
    atomicAdd(penalty, (d0 - 1.0f) * (d0 - 1.0f) + (d1 - 1.0f) * (d1 - 1.0f));
  }
}

// ---------------------------------------------------------------------------
// Power norm helpers
// ---------------------------------------------------------------------------
__global__ __launch_bounds__(256) void ssq_kernel(
    const float* __restrict__ p, float* __restrict__ ssq, int n) {
  float s = 0.0f;
  for (long i = (long)blockIdx.x * 256 + threadIdx.x; i < n; i += (long)gridDim.x * 256) {
    const float v = p[i]; s += v * v;
  }
  for (int o = 16; o > 0; o >>= 1) s += __shfl_down(s, o, 32);
  __shared__ float sh[8];
  const int wave = threadIdx.x >> 5, lane = threadIdx.x & 31;
  if (lane == 0) sh[wave] = s;
  __syncthreads();
  if (threadIdx.x == 0) {
    float t = 0.0f;
    for (int w = 0; w < 8; ++w) t += sh[w];
    atomicAdd(ssq, t);
  }
}

__global__ __launch_bounds__(128) void scale_z_kernel(
    const float* __restrict__ pos, const float* __restrict__ ssq,
    float* __restrict__ z, bf_t* __restrict__ zb) {
  const float power = sqrtf(*ssq / (float)(NG * H));
  const float s = power > 1.0f ? 1.0f / power : 1.0f;
  const long i = (long)blockIdx.x * H + threadIdx.x;
  const float v = pos[i] * s;
  z[i] = v;
  zb[i] = (bf_t)v;
}

__global__ __launch_bounds__(128) void gscale_kernel(
    const float* __restrict__ colsum, float* __restrict__ gvec) {
  __shared__ float red[128];
  const int f = threadIdx.x;
  const float cm = colsum[f] / (float)NTOT;
  red[f] = cm * cm;
  __syncthreads();
  for (int o = 64; o > 0; o >>= 1) {
    if (f < o) red[f] += red[f + o];
    __syncthreads();
  }
  const float power = sqrtf(red[0] / (float)H);
  const float s = power > 1.0f ? 1.0f / power : 1.0f;
  gvec[f] = cm * s;
}

__global__ __launch_bounds__(128) void gemb_kernel(
    const float* __restrict__ gvec, float* __restrict__ out) {
  out[(long)blockIdx.x * H + threadIdx.x] = gvec[threadIdx.x];
}

// ---------------------------------------------------------------------------
// VQ: argmin_k ( ||cb_k||^2 - 2 z.cb_k )  (||z||^2 constant per row)
// ---------------------------------------------------------------------------
__global__ __launch_bounds__(128) void cbsq_kernel(
    const float* __restrict__ cb, float* __restrict__ cbsq) {
  const int k = threadIdx.x;
  float s = 0.0f;
  for (int f = 0; f < H; ++f) { const float v = cb[(long)k * H + f]; s += v * v; }
  cbsq[k] = s;
}

__global__ __launch_bounds__(128) void argmin_kernel(
    const float* __restrict__ S, const float* __restrict__ cbsq,
    int* __restrict__ minidx) {
  __shared__ float sv[128];
  __shared__ int si[128];
  const int g = blockIdx.x, k = threadIdx.x;
  sv[k] = cbsq[k] - 2.0f * S[(long)g * KCB + k];
  si[k] = k;
  __syncthreads();
  for (int o = 64; o > 0; o >>= 1) {
    if (k < o) {
      if (sv[k + o] < sv[k] || (sv[k + o] == sv[k] && si[k + o] < si[k])) {
        sv[k] = sv[k + o]; si[k] = si[k + o];
      }
    }
    __syncthreads();
  }
  if (k == 0) minidx[g] = si[0];
}

// ---------------------------------------------------------------------------
// Deterministic transmission: threefry2x32(key(42)) uniforms, JAX semantics
// ---------------------------------------------------------------------------
__device__ __forceinline__ unsigned rotl32(unsigned x, int n) {
  return (x << n) | (x >> (32 - n));
}

__global__ __launch_bounds__(256) void transmit_kernel(
    const int* __restrict__ minidx, const float* __restrict__ cp_ptr,
    int* __restrict__ qidx) {
  const int g = blockIdx.x * 256 + threadIdx.x;
  if (g >= NG) return;
  const unsigned k0 = 0u, k1 = 42u, k2 = k0 ^ k1 ^ 0x1BD11BDAu;
  const unsigned ks[3] = {k0, k1, k2};
  const unsigned i = (g < 256) ? (unsigned)g : (unsigned)(g - 256);
  unsigned x0 = i + ks[0];
  unsigned x1 = (i + 256u) + ks[1];
  const int rot[2][4] = {{13, 15, 26, 6}, {17, 29, 16, 24}};
#pragma unroll
  for (int r = 0; r < 5; ++r) {
    const int* rr = rot[r & 1];
#pragma unroll
    for (int j = 0; j < 4; ++j) { x0 += x1; x1 = rotl32(x1, rr[j]); x1 ^= x0; }
    x0 += ks[(r + 1) % 3];
    x1 += ks[(r + 2) % 3] + (unsigned)(r + 1);
  }
  const unsigned bits = (g < 256) ? x0 : x1;
  const float p = __uint_as_float(0x3f800000u | (bits >> 9)) - 1.0f;
  const float cp = *cp_ptr;
  int srow;
  if (p <= cp) {
    srow = g;
  } else {
    const int integer = (int)floorf((p - cp) / ((1.0f - cp) / (float)KCB));
    srow = ((g + 1 + integer) % NG + NG) % NG;
  }
  qidx[g] = minidx[srow];
}

__global__ __launch_bounds__(128) void quantcopy_kernel(
    const float* __restrict__ cb, const int* __restrict__ qidx,
    float* __restrict__ qout, bf_t* __restrict__ qb) {
  const int g = blockIdx.x, f = threadIdx.x;
  const float v = cb[(long)qidx[g] * H + f];
  qout[(long)g * H + f] = v;
  qb[(long)g * H + f] = (bf_t)v;
}

// ---------------------------------------------------------------------------
// Classifier tail: logits = h1 @ l2W + l2b, then log_softmax
// ---------------------------------------------------------------------------
__global__ __launch_bounds__(128) void logits_kernel(
    const bf_t* __restrict__ h1, const float* __restrict__ W,
    const float* __restrict__ b, float* __restrict__ out) {
  __shared__ float red[128];
  __shared__ float lg[NCLS];
  const int g = blockIdx.x, f = threadIdx.x;
  const float hv = (float)h1[(long)g * H + f];
  for (int c = 0; c < NCLS; ++c) {
    red[f] = hv * W[f * NCLS + c];
    __syncthreads();
    for (int o = 64; o > 0; o >>= 1) {
      if (f < o) red[f] += red[f + o];
      __syncthreads();
    }
    if (f == 0) lg[c] = red[0] + b[c];
    __syncthreads();
  }
  if (f == 0) {
    float mx = lg[0];
    for (int c = 1; c < NCLS; ++c) mx = fmaxf(mx, lg[c]);
    float se = 0.0f;
    for (int c = 0; c < NCLS; ++c) se += __expf(lg[c] - mx);
    const float lse = mx + __logf(se);
    for (int c = 0; c < NCLS; ++c) out[(long)g * NCLS + c] = lg[c] - lse;
  }
}

__global__ void zero_kernel(float* colsum, float* penalty, float* ssq) {
  const int t = threadIdx.x;
  if (t < H) colsum[t] = 0.0f;
  if (t == 0) { penalty[0] = 0.0f; ssq[0] = 0.0f; }
}

__global__ void finalize_kernel(const float* penalty, float* out) {
  out[0] = penalty[0] / (float)(NG * 2);
}

// ---------------------------------------------------------------------------
// Host launcher
// ---------------------------------------------------------------------------
extern "C" void kernel_launch(void* const* d_in, const int* in_sizes, int n_in,
                              void* d_out, int out_size, void* d_ws, size_t ws_size,
                              hipStream_t stream) {
  // Input indices (setup_inputs() insertion order, params depth-first)
  const float* x  = (const float*)d_in[0];
  const int*   ei = (const int*)d_in[1];
  // d_in[2] = batch (unused: graphs contiguous, 128 nodes each)
  const float* cW1[3] = {(const float*)d_in[3],  (const float*)d_in[9],  (const float*)d_in[15]};
  const float* cb1[3] = {(const float*)d_in[4],  (const float*)d_in[10], (const float*)d_in[16]};
  const float* cW2[3] = {(const float*)d_in[5],  (const float*)d_in[11], (const float*)d_in[17]};
  const float* cb2[3] = {(const float*)d_in[6],  (const float*)d_in[12], (const float*)d_in[18]};
  const float* cg [3] = {(const float*)d_in[7],  (const float*)d_in[13], (const float*)d_in[19]};
  const float* cbt[3] = {(const float*)d_in[8],  (const float*)d_in[14], (const float*)d_in[20]};
  const float* c1W = (const float*)d_in[21];
  const float* c1b = (const float*)d_in[22];
  const float* c2W = (const float*)d_in[23];
  const float* c2b = (const float*)d_in[24];
  const float* l1W = (const float*)d_in[25];
  const float* l1b = (const float*)d_in[26];
  const float* l2W = (const float*)d_in[27];
  const float* l2b = (const float*)d_in[28];
  const float* cb  = (const float*)d_in[29];
  const float* cpp = (const float*)d_in[30];

  // Workspace layout
  char* ws = (char*)d_ws;
  size_t off = 0;
  auto alloc = [&](size_t bytes) { void* p = ws + off; off += (bytes + 255) & ~(size_t)255; return p; };
  bf_t*  tb     = (bf_t*)alloc((size_t)NTOT * H * 2);   // aggregated (bf16)
  bf_t*  ub     = (bf_t*)alloc((size_t)NTOT * H * 2);   // hidden (bf16)
  float* vf     = (float*)alloc((size_t)NTOT * H * 4);  // pre-BN / m
  float* hf     = (float*)alloc((size_t)NTOT * H * 4);  // post-BN h
  bf_t*  hb     = (bf_t*)alloc((size_t)NTOT * H * 2);   // post-BN h (bf16)
  float* a      = (float*)alloc((size_t)NTOT * 2 * 4);
  float* pos    = (float*)alloc((size_t)NG * H * 4);
  float* z      = (float*)alloc((size_t)NG * H * 4);
  bf_t*  zb     = (bf_t*)alloc((size_t)NG * H * 2);
  float* S      = (float*)alloc((size_t)NG * KCB * 4);
  bf_t*  qb     = (bf_t*)alloc((size_t)NG * H * 2);
  bf_t*  h1b    = (bf_t*)alloc((size_t)NG * H * 2);
  float* colsum = (float*)alloc(H * 4);
  float* mean   = (float*)alloc(H * 4);
  float* rstd   = (float*)alloc(H * 4);
  float* cbsq   = (float*)alloc(KCB * 4);
  float* gvec   = (float*)alloc(H * 4);
  float* pen    = (float*)alloc(4);
  float* ssq    = (float*)alloc(4);
  int*   minidx = (int*)alloc(NG * 4);
  int*   qidx   = (int*)alloc(NG * 4);

  float* out_ls   = (float*)d_out;                 // [NG, 10]
  float* out_q    = out_ls + (size_t)NG * NCLS;    // [NG, H]
  float* out_ge   = out_q + (size_t)NG * H;        // [NG, H]
  float* out_pen  = out_ge + (size_t)NG * H;       // [1]

  zero_kernel<<<1, 256, 0, stream>>>(colsum, pen, ssq);

  // --- 3 GIN conv layers ---
  const float* hcur = x;
  for (int ly = 0; ly < 3; ++ly) {
    aggregate_kernel<<<NG, 256, 0, stream>>>(hcur, ei, tb);
    // u = relu(t @ W1 + b1) -> bf16 only
    gemm128_kernel<1, false, true, false, true>
        <<<NTOT / 128, 256, 0, stream>>>(tb, cW1[ly], cb1[ly], nullptr, ub);
    // v = relu(u @ W2 + b2) -> f32 only (for BN)
    gemm128_kernel<1, false, true, true, false>
        <<<NTOT / 128, 256, 0, stream>>>(ub, cW2[ly], cb2[ly], vf, nullptr);
    bn_stats_kernel<<<H, 256, 0, stream>>>(vf, mean, rstd);
    bn_apply_kernel<<<(NTOT * H) / 256, 256, 0, stream>>>(vf, mean, rstd, cg[ly], cbt[ly], hf, hb);
    hcur = hf;
  }

  // --- assignment head: m = tanh(h @ c1W + c1b) -> f32 only ---
  gemm128_kernel<2, false, true, true, false>
      <<<NTOT / 128, 256, 0, stream>>>(hb, c1W, c1b, vf, nullptr);
  assign_kernel<<<NTOT / 8, 256, 0, stream>>>(vf, c2W, c2b, a);

  // --- per-graph pooling + adjacency penalty ---
  graph_kernel<<<NG, 128, 0, stream>>>(hf, a, pos, colsum);
  adj_kernel<<<NG, 256, 0, stream>>>(a, ei, pen);

  // --- power norms ---
  ssq_kernel<<<64, 256, 0, stream>>>(pos, ssq, NG * H);
  scale_z_kernel<<<NG, 128, 0, stream>>>(pos, ssq, z, zb);
  gscale_kernel<<<1, 128, 0, stream>>>(colsum, gvec);
  gemb_kernel<<<NG, 128, 0, stream>>>(gvec, out_ge);

  // --- VQ nearest-codeword via WMMA GEMM (S = z @ cb^T, no bias) ---
  cbsq_kernel<<<1, 128, 0, stream>>>(cb, cbsq);
  gemm128_kernel<0, true, false, true, false>
      <<<NG / 128, 256, 0, stream>>>(zb, cb, nullptr, S, nullptr);
  argmin_kernel<<<NG, 128, 0, stream>>>(S, cbsq, minidx);

  // --- stochastic transmission (deterministic threefry) + quantize ---
  transmit_kernel<<<(NG + 255) / 256, 256, 0, stream>>>(minidx, cpp, qidx);
  quantcopy_kernel<<<NG, 128, 0, stream>>>(cb, qidx, out_q, qb);

  // --- classifier head: h1 = relu(q @ l1W + l1b) -> bf16 only ---
  gemm128_kernel<1, false, true, false, true>
      <<<NG / 128, 256, 0, stream>>>(qb, l1W, l1b, nullptr, h1b);
  logits_kernel<<<NG, 128, 0, stream>>>(h1b, l2W, l2b, out_ls);

  finalize_kernel<<<1, 1, 0, stream>>>(pen, out_pen);
  (void)in_sizes; (void)n_in; (void)out_size; (void)ws_size;
}
